// my_attention_7404523618904
// MI455X (gfx1250) — matrix operand order repaired and checked by reading, not compiled, not run on hardware
//
#include <hip/hip_runtime.h>
#include <hip/hip_bf16.h>

// ---------------------------------------------------------------------------
// MI455X (gfx1250) causal MHA: B=8, N=2048, D=256, HEAD=8, ATT=32
// 3x WMMA projection GEMMs -> transposed flash attention (S^T = K*Q^T,
// O^T = V^T*P^T, per-lane softmax) -> WMMA output GEMM.
// All matmuls: v_wmma_f32_16x16x32_bf16 (f32 accumulate).
// ---------------------------------------------------------------------------

typedef __attribute__((ext_vector_type(16))) __bf16 v16bf;
typedef __attribute__((ext_vector_type(8)))  __bf16 v8bf;
typedef __attribute__((ext_vector_type(4)))  __bf16 v4bf;
typedef __attribute__((ext_vector_type(8)))  float   v8f;

#define NEG_INF (-__builtin_inff())

#define WMMA_BF16(a, b, c) \
  __builtin_amdgcn_wmma_f32_16x16x32_bf16(false, (a), false, (b), (short)0, (c), false, false)

static __device__ __forceinline__ v16bf combine16(v8bf lo, v8bf hi) {
  v16bf r;
#pragma unroll
  for (int i = 0; i < 8; ++i) { r[i] = lo[i]; r[8 + i] = hi[i]; }
  return r;
}

static __device__ __forceinline__ v8f zero8() {
  return (v8f){0.f, 0.f, 0.f, 0.f, 0.f, 0.f, 0.f, 0.f};
}

// ---------------------------------------------------------------------------
// GEMM: y[m][c] = (sum_k A[m][k] * W[c][k] + bias[c]) * scale
//   M = 16384, K = 256, Ncol = 256. Block 64x64, 4 waves 2x2, wave = 2x2 WMMA.
// MODE 0: A f32, out bf16 q/k layout [head*8+batch][n][dim]
// MODE 2: A f32, out bf16 vT layout  [head*8+batch][dim][n]
// MODE 3: A bf16, out f32 row-major  [m][c]
// ---------------------------------------------------------------------------
template <int MODE>
__global__ __launch_bounds__(128) void gemm256(const float* __restrict__ A32,
                                               const __bf16* __restrict__ A16,
                                               const float* __restrict__ W,
                                               const float* __restrict__ bias,
                                               __bf16* __restrict__ out16,
                                               float* __restrict__ out32,
                                               float scale) {
  __shared__ __bf16 As[64][32];
  __shared__ __bf16 Bs[64][32];

  const int tid  = threadIdx.x;
  const int lane = tid & 31;
  const int wave = tid >> 5;
  const int wm = (wave >> 1) * 32;
  const int wn = (wave & 1) * 32;
  const int mbase = blockIdx.x * 64;
  const int nbase = blockIdx.y * 64;

  const int lr   = tid & 63;
  const int lseg = (tid >> 6) * 16;

  v8f acc[2][2];
#pragma unroll
  for (int i = 0; i < 2; ++i)
#pragma unroll
    for (int j = 0; j < 2; ++j) acc[i][j] = zero8();

  for (int k0 = 0; k0 < 256; k0 += 32) {
    if (MODE == 3) {
      const __bf16* src = A16 + (size_t)(mbase + lr) * 256 + k0 + lseg;
      *(v8bf*)&As[lr][lseg]     = *(const v8bf*)(src);
      *(v8bf*)&As[lr][lseg + 8] = *(const v8bf*)(src + 8);
    } else {
      const float* src = A32 + (size_t)(mbase + lr) * 256 + k0 + lseg;
#pragma unroll
      for (int j = 0; j < 4; ++j) {
        float4 f = *(const float4*)(src + 4 * j);
        v4bf hh;
        hh[0] = (__bf16)f.x; hh[1] = (__bf16)f.y; hh[2] = (__bf16)f.z; hh[3] = (__bf16)f.w;
        *(v4bf*)&As[lr][lseg + 4 * j] = hh;
      }
    }
    {
      const float* src = W + (size_t)(nbase + lr) * 256 + k0 + lseg;
#pragma unroll
      for (int j = 0; j < 4; ++j) {
        float4 f = *(const float4*)(src + 4 * j);
        v4bf hh;
        hh[0] = (__bf16)f.x; hh[1] = (__bf16)f.y; hh[2] = (__bf16)f.z; hh[3] = (__bf16)f.w;
        *(v4bf*)&Bs[lr][lseg + 4 * j] = hh;
      }
    }
    __syncthreads();

    const int r = lane & 15;
    const int h = lane >> 4;
    v16bf a[2], b[2];
#pragma unroll
    for (int s = 0; s < 2; ++s) {
      const __bf16* ap = &As[wm + s * 16 + r][0];
      a[s] = combine16(*(const v8bf*)(ap + h * 8), *(const v8bf*)(ap + 16 + h * 8));
      const __bf16* bp = &Bs[wn + s * 16 + r][0];
      b[s] = combine16(*(const v8bf*)(bp + h * 8), *(const v8bf*)(bp + 16 + h * 8));
    }
#pragma unroll
    for (int i = 0; i < 2; ++i)
#pragma unroll
      for (int j = 0; j < 2; ++j)
        acc[i][j] = WMMA_BF16(a[i], b[j], acc[i][j]);
    __syncthreads();
  }

  const int cL     = lane & 15;
  const int rowOff = (lane >> 4) * 8;
#pragma unroll
  for (int i = 0; i < 2; ++i) {
#pragma unroll
    for (int j = 0; j < 2; ++j) {
      const int col = nbase + wn + j * 16 + cL;
      const float bv = bias[col];
#pragma unroll
      for (int e = 0; e < 8; ++e) {
        const int row = mbase + wm + i * 16 + rowOff + e;
        const float val = (acc[i][j][e] + bv) * scale;
        if (MODE == 3) {
          out32[(size_t)row * 256 + col] = val;
        } else {
          const int head = col >> 5, dim = col & 31;
          const int batch = row >> 11, n = row & 2047;
          const int bh = head * 8 + batch;
          if (MODE == 2)
            out16[((size_t)bh * 32 + dim) * 2048 + n] = (__bf16)val;
          else
            out16[((size_t)bh * 2048 + n) * 32 + dim] = (__bf16)val;
        }
      }
    }
  }
}

// ---------------------------------------------------------------------------
// Transposed flash step over one 32-key tile.
//   S^T tile t = WMMA(K_tile_t (A), Q^T (B))  -> lane owns one query column
//   softmax stats: per-lane over 16 regs + one shfl_xor(16) per stat
//   O^T += WMMA(V^T tile (A), P^T (B, via 2x b128 LDS round trip))
// ---------------------------------------------------------------------------
template <bool MASKED>
static __device__ __forceinline__ void flash_step(
    int kbase, int query, int c, int h, int rowOff,
    const __bf16* __restrict__ kbh, const __bf16* __restrict__ vbh,
    const v16bf& qbf, __bf16* __restrict__ pslab,
    v8f& oT0, v8f& oT1, float& m, float& l) {
  // K A-fragments: row = key, K-dim = 32 head dims (contiguous in memory)
  const __bf16* k0 = kbh + (size_t)(kbase + c) * 32;
  const __bf16* k1 = kbh + (size_t)(kbase + 16 + c) * 32;
  const v16bf ka0 = combine16(*(const v8bf*)(k0 + h * 8), *(const v8bf*)(k0 + 16 + h * 8));
  const v16bf ka1 = combine16(*(const v8bf*)(k1 + h * 8), *(const v8bf*)(k1 + 16 + h * 8));

  v8f s0 = WMMA_BF16(ka0, qbf, zero8());   // keys kbase+rowOff+i, query col c
  v8f s1 = WMMA_BF16(ka1, qbf, zero8());   // keys kbase+16+rowOff+i

  float x0[8], x1[8];
  float mx = NEG_INF;
#pragma unroll
  for (int i = 0; i < 8; ++i) {
    if (MASKED) {
      const int key = kbase + rowOff + i;
      x0[i] = (key      <= query) ? s0[i] : NEG_INF;
      x1[i] = (key + 16 <= query) ? s1[i] : NEG_INF;
    } else {
      x0[i] = s0[i];
      x1[i] = s1[i];
    }
    mx = fmaxf(mx, fmaxf(x0[i], x1[i]));
  }
  mx = fmaxf(mx, __shfl_xor(mx, 16, 32));   // combine the two key-row halves
  const float mnew = fmaxf(m, mx);

  float sum = 0.f;
  v8bf p0, p1;
#pragma unroll
  for (int i = 0; i < 8; ++i) {
    const float e0 = __expf(x0[i] - mnew);
    const float e1 = __expf(x1[i] - mnew);
    sum += e0 + e1;
    p0[i] = (__bf16)e0;
    p1[i] = (__bf16)e1;
  }
  sum += __shfl_xor(sum, 16, 32);
  const float fac = __expf(m - mnew);       // first tile: exp(-inf)=0
  l = l * fac + sum;
  m = mnew;
#pragma unroll
  for (int i = 0; i < 8; ++i) { oT0[i] *= fac; oT1[i] *= fac; }

  // P stored as P[query][key_rel] (two contiguous 16B stores per lane)
  *(v8bf*)(pslab + c * 32 + rowOff)      = p0;
  *(v8bf*)(pslab + c * 32 + 16 + rowOff) = p1;
  __asm volatile("s_wait_dscnt 0" ::: "memory");   // intra-wave LDS RAW

  // P^T B-fragment: lane -> query c, keys h*8..h*8+7 and 16+h*8..16+h*8+7
  const __bf16* pp = pslab + c * 32 + h * 8;
  const v16bf pb = combine16(*(const v8bf*)pp, *(const v8bf*)(pp + 16));

  // V^T A-fragments: row = head dim, K-dim = 32 keys (contiguous: V transposed)
  const __bf16* v0 = vbh + (size_t)c * 2048 + kbase;
  const __bf16* v1 = vbh + (size_t)(16 + c) * 2048 + kbase;
  const v16bf va0 = combine16(*(const v8bf*)(v0 + h * 8), *(const v8bf*)(v0 + 16 + h * 8));
  const v16bf va1 = combine16(*(const v8bf*)(v1 + h * 8), *(const v8bf*)(v1 + 16 + h * 8));

  oT0 = WMMA_BF16(va0, pb, oT0);   // dims 0..15   x queries
  oT1 = WMMA_BF16(va1, pb, oT1);   // dims 16..31  x queries
  __asm volatile("" ::: "memory"); // keep this iter's LDS reads before next stores
}

__global__ __launch_bounds__(128) void attn_flash(const __bf16* __restrict__ q,
                                                  const __bf16* __restrict__ kmat,
                                                  const __bf16* __restrict__ vT,
                                                  __bf16* __restrict__ attn_out) {
  __shared__ __bf16 Ps[4][16][32];   // per-wave P^T slab

  const int tid  = threadIdx.x;
  const int lane = tid & 31;
  const int wave = tid >> 5;
  const int bh    = blockIdx.y;                  // head*8 + batch
  const int qtile = blockIdx.x * 64 + wave * 16;
  const int c = lane & 15;                       // query column / A-frag row
  const int h = lane >> 4;
  const int rowOff = h * 8;
  const int query = qtile + c;                   // this lane's query

  // Q^T B-fragment (loaded once; q was pre-scaled by 1/sqrt(32) in its GEMM)
  const __bf16* qp = q + ((size_t)bh * 2048 + query) * 32 + h * 8;
  const v16bf qbf = combine16(*(const v8bf*)qp, *(const v8bf*)(qp + 16));

  const __bf16* kbh = kmat + (size_t)bh * 2048 * 32;
  const __bf16* vbh = vT + (size_t)bh * 32 * 2048;
  __bf16* pslab = &Ps[wave][0][0];

  v8f oT0 = zero8(), oT1 = zero8();
  float m = NEG_INF, l = 0.f;

  int kbase = 0;
  // full tiles: every key <= every query in this wave -> no masking at all
  for (; kbase + 31 <= qtile; kbase += 32)
    flash_step<false>(kbase, query, c, h, rowOff, kbh, vbh, qbf, pslab, oT0, oT1, m, l);
  // exactly one diagonal tile (stray over-reads hit finite bf16 inside ws and
  // are multiplied by exactly-zero P entries)
  flash_step<true>(kbase, query, c, h, rowOff, kbh, vbh, qbf, pslab, oT0, oT1, m, l);

  // O^T: lane holds query `query`, dims rowOff..rowOff+7 (+16 for oT1)
  const int head  = bh >> 3;
  const int batch = bh & 7;
  const float inv = 1.0f / l;
  v8bf r0, r1;
#pragma unroll
  for (int i = 0; i < 8; ++i) {
    r0[i] = (__bf16)(oT0[i] * inv);
    r1[i] = (__bf16)(oT1[i] * inv);
  }
  __bf16* dst = attn_out + ((size_t)(batch * 2048 + query)) * 256 + head * 32;
  *(v8bf*)(dst + rowOff)      = r0;   // contiguous 16B stores
  *(v8bf*)(dst + 16 + rowOff) = r1;
}

// ---------------------------------------------------------------------------
extern "C" void kernel_launch(void* const* d_in, const int* in_sizes, int n_in,
                              void* d_out, int out_size, void* d_ws, size_t ws_size,
                              hipStream_t stream) {
  const float* query = (const float*)d_in[0];
  const float* key   = (const float*)d_in[1];
  const float* value = (const float*)d_in[2];
  const float* Wq = (const float*)d_in[3];
  const float* bq = (const float*)d_in[4];
  const float* Wk = (const float*)d_in[5];
  const float* bk = (const float*)d_in[6];
  const float* Wv = (const float*)d_in[7];
  const float* bv = (const float*)d_in[8];
  const float* Wo = (const float*)d_in[9];
  const float* bo = (const float*)d_in[10];
  float* out = (float*)d_out;

  char* ws = (char*)d_ws;
  const size_t qkv_bytes = (size_t)64 * 2048 * 32 * sizeof(__bf16);   // 8 MB each
  __bf16* qb    = (__bf16*)(ws);
  __bf16* kb    = (__bf16*)(ws + qkv_bytes);
  __bf16* vTb   = (__bf16*)(ws + 2 * qkv_bytes);
  __bf16* attnb = (__bf16*)(ws + 3 * qkv_bytes);

  const float qscale = 0.17677669529663687f;   // 1/sqrt(32), folded into q
  const dim3 gblk(256, 4), gthr(128);
  gemm256<0><<<gblk, gthr, 0, stream>>>(query, nullptr, Wq, bq, qb,  nullptr, qscale);
  gemm256<0><<<gblk, gthr, 0, stream>>>(key,   nullptr, Wk, bk, kb,  nullptr, 1.0f);
  gemm256<2><<<gblk, gthr, 0, stream>>>(value, nullptr, Wv, bv, vTb, nullptr, 1.0f);
  attn_flash<<<dim3(32, 64), gthr, 0, stream>>>(qb, kb, vTb, attnb);
  gemm256<3><<<gblk, gthr, 0, stream>>>(nullptr, attnb, Wo, bo, nullptr, out, 1.0f);
}
